// LinearCrossAttention_37967510896792
// MI455X (gfx1250) — compile-verified, matrix operand order checked by reading, not run on hardware
//
#include <hip/hip_runtime.h>

// ---------------------------------------------------------------------------
// LinearCrossAttention on MI455X (gfx1250): all GEMMs via v_wmma_f32_16x16x32_bf16
// f32 inputs are converted (RNE) to bf16 on the LDS staging path; accumulate f32.
// Big GEMMs double-buffer LDS tiles; pure-bf16 copies use GLOBAL_LOAD_ASYNC_TO_LDS.
// ---------------------------------------------------------------------------

typedef unsigned short u16;
typedef unsigned int   u32;
typedef __attribute__((ext_vector_type(16))) __bf16 v16bf;
typedef __attribute__((ext_vector_type(8)))  float  v8f;

__device__ __forceinline__ u16 f2bf(float x) {
  u32 u = __float_as_uint(x);
  u = (u + 0x7FFFu + ((u >> 16) & 1u)) >> 16;   // round-to-nearest-even
  return (u16)u;
}
__device__ __forceinline__ float bf2f(u16 h) {
  return __uint_as_float(((u32)h) << 16);
}
__device__ __forceinline__ float elu1(float x) {
  return x > 0.0f ? x + 1.0f : __expf(x);        // elu(x)+1
}

// Async global->LDS copies (ASYNCcnt path). LDS dest must be 16B aligned.
__device__ __forceinline__ void async_cp_32B(u32 lds, const u16* g) {
  asm volatile("global_load_async_to_lds_b128 %0, %1, off\n\t"
               "global_load_async_to_lds_b128 %0, %1, off offset:16"
               :: "v"(lds), "v"(g) : "memory");
}
__device__ __forceinline__ void async_cp_64B(u32 lds, const u16* g) {
  asm volatile("global_load_async_to_lds_b128 %0, %1, off\n\t"
               "global_load_async_to_lds_b128 %0, %1, off offset:16\n\t"
               "global_load_async_to_lds_b128 %0, %1, off offset:32\n\t"
               "global_load_async_to_lds_b128 %0, %1, off offset:48"
               :: "v"(lds), "v"(g) : "memory");
}
__device__ __forceinline__ void wait_async0() {
  asm volatile("s_wait_asynccnt 0" ::: "memory");
}
__device__ __forceinline__ u32 lds_off(const void* p) {   // LDS byte offset
  return (u32)(__UINTPTR_TYPE__)p;
}

// Build the per-lane 16x32 bf16 WMMA operand from LDS.
// ISA 16-bit A layout: lanes 0-15 hold M=lane; VGPR v<4 -> K=half*8+2v{,+1},
// v>=4 -> K=16+half*8+2(v-4){,+1}. Row stride must be even (keeps dword align).
__device__ __forceinline__ v16bf lds_frag(const u16* p, int row_base, int stride,
                                          int k_off, int lane) {
  const u16* row = p + (row_base + (lane & 15)) * stride + k_off;
  const int half8 = ((lane >> 4) & 1) << 3;
  union { v16bf v; u32 u[8]; } f;
#pragma unroll
  for (int r = 0; r < 8; ++r) {
    int k = ((r & 4) ? 16 : 0) + half8 + ((r & 3) << 1);
    f.u[r] = *(const u32*)(row + k);
  }
  return f.v;
}

__device__ __forceinline__ v8f wmma_bf16(v16bf a, v16bf b, v8f c) {
  return __builtin_amdgcn_wmma_f32_16x16x32_bf16(false, a, false, b,
                                                 (short)0, c, false, false);
}

// ---------------------------------------------------------------------------
// Generic K=384 GEMM:  Y[Nrows,Ncols] = act(X[Nrows,384] @ W[384,Ncols]) (+bias)
// ACT: 0=none, 1=elu+1 on all cols, 2=elu+1 on cols<384 (K half of KV proj)
// Tile 128x64, BK=32, 256 threads = 8 waves (4x2), wave = 32x32 = 2x2 frags.
// Double-buffered LDS; A staged via async-to-LDS when already bf16.
// ---------------------------------------------------------------------------
template <bool ABF16>
__device__ __forceinline__ void stage_tiles(const void* __restrict__ Xv,
                                            const float* __restrict__ W,
                                            int row0, int col0, int kk, int tid,
                                            int Ncols, u16* As, u16* BsT) {
  // ---- A tile (128x32) bf16, row-major, stride 40 ----
  const int r = tid >> 1;
  const int c = (tid & 1) << 4;            // 16 elems (32B) per thread
  if (ABF16) {
    const u16* src = (const u16*)Xv + (size_t)(row0 + r) * 384 + kk + c;
    async_cp_32B(lds_off(As + r * 40 + c), src);
  } else {
    const float* src = (const float*)Xv + (size_t)(row0 + r) * 384 + kk + c;
    u32* dst = (u32*)(As + r * 40 + c);
#pragma unroll
    for (int i = 0; i < 8; ++i)
      dst[i] = (u32)f2bf(src[2 * i]) | ((u32)f2bf(src[2 * i + 1]) << 16);
  }
  // ---- B tile (32x64) transposed as bf16, stride 40 ----
  const int kr = tid >> 3;                 // 0..31
  const int nb = (tid & 7) << 3;           // 8 cols per thread
  const float* wsrc = W + (size_t)(kk + kr) * Ncols + col0 + nb;
#pragma unroll
  for (int i = 0; i < 8; ++i) BsT[(nb + i) * 40 + kr] = f2bf(wsrc[i]);
  if (kk + 32 < 384) __builtin_prefetch(wsrc + 32 * (size_t)Ncols, 0, 1);
}

template <int ACT, bool ABF16, bool BIAS, bool OUTF32>
__global__ __launch_bounds__(256) void gemm384(const void* __restrict__ Xv,
                                               const float* __restrict__ W,
                                               const float* __restrict__ bias,
                                               void* __restrict__ Yv,
                                               int Nrows, int Ncols) {
  __shared__ __align__(16) u16 As[2][128 * 40];
  __shared__ __align__(16) u16 BsT[2][64 * 40];   // BsT[n][k]

  const int row0 = blockIdx.x * 128;
  const int col0 = blockIdx.y * 64;
  const int tid  = threadIdx.x;
  const int lane = tid & 31;
  const int wave = tid >> 5;
  const int wm   = wave & 3;     // 4 row-blocks of 32
  const int wn   = wave >> 2;    // 2 col-blocks of 32

  v8f acc[2][2] = {};

  stage_tiles<ABF16>(Xv, W, row0, col0, 0, tid, Ncols, As[0], BsT[0]);
  int p = 0;
  for (int kk = 0; kk < 384; kk += 32) {
    if (ABF16) wait_async0();      // my async stores into As[p] complete
    __syncthreads();               // everyone's staging of buffer p published
    if (kk + 32 < 384)
      stage_tiles<ABF16>(Xv, W, row0, col0, kk + 32, tid, Ncols,
                         As[p ^ 1], BsT[p ^ 1]);   // overlaps with WMMA below
    v16bf bfrag[2];
#pragma unroll
    for (int fn = 0; fn < 2; ++fn)
      bfrag[fn] = lds_frag(BsT[p], wn * 32 + fn * 16, 40, 0, lane);
#pragma unroll
    for (int fm = 0; fm < 2; ++fm) {
      v16bf a = lds_frag(As[p], wm * 32 + fm * 16, 40, 0, lane);
#pragma unroll
      for (int fn = 0; fn < 2; ++fn)
        acc[fm][fn] = wmma_bf16(a, bfrag[fn], acc[fm][fn]);
    }
    p ^= 1;
  }

  // ---- epilogue: C layout = lane n, VGPR r -> row r + 8*(lane>=16) ----
#pragma unroll
  for (int fm = 0; fm < 2; ++fm)
#pragma unroll
    for (int fn = 0; fn < 2; ++fn)
#pragma unroll
      for (int r = 0; r < 8; ++r) {
        const int row = wm * 32 + fm * 16 + ((lane >> 4) << 3) + r;
        const int col = wn * 32 + fn * 16 + (lane & 15);
        const int gc  = col0 + col;
        float v = acc[fm][fn][r];
        if (ACT == 1 || (ACT == 2 && gc < 384)) v = elu1(v);
        if (BIAS) v += bias[gc];
        const size_t idx = (size_t)(row0 + row) * Ncols + gc;
        if (OUTF32) ((float*)Yv)[idx] = v;
        else        ((u16*)Yv)[idx]   = f2bf(v);
      }
}

// ---------------------------------------------------------------------------
// Stage 3: per (b,h): kv[d,m] = sum_n k[n,d]*v[n,m]  (64x64, f32) + ksum[d].
// 4 waves, each owns an n-chunk of 1024; per-wave transposed LDS staging so
// K^T is a native WMMA A operand; LDS-atomic cross-wave reduction.
// ---------------------------------------------------------------------------
__global__ __launch_bounds__(128) void kv_outer(const u16* __restrict__ kvp,
                                                float* __restrict__ kv_mat,
                                                float* __restrict__ ksum) {
  __shared__ u16  Ks[4][64 * 34];   // Ks[w][d][n] = k[n,d]
  __shared__ u16  Vs[4][64 * 34];   // Vs[w][m][n] = v[n,m]
  __shared__ float red[64 * 64];
  __shared__ float kpart[128];

  const int bh = blockIdx.x;
  const int b  = bh / 6, h = bh % 6;
  const int tid = threadIdx.x, lane = tid & 31, wave = tid >> 5;

  for (int i = tid; i < 64 * 64; i += 128) red[i] = 0.0f;
  __syncthreads();

  const size_t baseK = ((size_t)b * 4096) * 768 + h * 64;
  const size_t baseV = baseK + 384;

  v8f acc[4][4] = {};
  u16* ks = Ks[wave];
  u16* vs = Vs[wave];

  for (int step = 0; step < 32; ++step) {
    const int n = wave * 1024 + step * 32 + lane;   // lane = n-row of the tile
    const u32* kp = (const u32*)(kvp + baseK + (size_t)n * 768);
    const u32* vp = (const u32*)(kvp + baseV + (size_t)n * 768);
#pragma unroll
    for (int i = 0; i < 32; ++i) {       // 64 d's as 32 dwords, transpose on store
      const u32 ku = kp[i], vu = vp[i];
      const int d0 = 2 * i;
      ks[d0 * 34 + lane]       = (u16)(ku & 0xffffu);
      ks[(d0 + 1) * 34 + lane] = (u16)(ku >> 16);
      vs[d0 * 34 + lane]       = (u16)(vu & 0xffffu);
      vs[(d0 + 1) * 34 + lane] = (u16)(vu >> 16);
    }
    __syncthreads();

    v16bf bfrag[4];
#pragma unroll
    for (int j = 0; j < 4; ++j) bfrag[j] = lds_frag(vs, j * 16, 34, 0, lane);
#pragma unroll
    for (int fd = 0; fd < 4; ++fd) {
      v16bf a = lds_frag(ks, fd * 16, 34, 0, lane);
#pragma unroll
      for (int j = 0; j < 4; ++j)
        acc[fd][j] = wmma_bf16(a, bfrag[j], acc[fd][j]);
    }
    __syncthreads();
  }

  // cross-wave reduction in LDS
#pragma unroll
  for (int fd = 0; fd < 4; ++fd)
#pragma unroll
    for (int j = 0; j < 4; ++j)
#pragma unroll
      for (int r = 0; r < 8; ++r) {
        const int d = fd * 16 + ((lane >> 4) << 3) + r;
        const int m = j * 16 + (lane & 15);
        atomicAdd(&red[d * 64 + m], acc[fd][j][r]);
      }
  __syncthreads();

  float* kvout = kv_mat + (size_t)bh * 4096;
  for (int i = tid; i < 4096; i += 128) kvout[i] = red[i];

  // ksum[d] = sum_n k[n,d]
  {
    const int d = tid & 63, part = tid >> 6;     // 2 partials per d
    float s = 0.0f;
    for (int n = part * 2048; n < part * 2048 + 2048; ++n)
      s += bf2f(kvp[baseK + (size_t)n * 768 + d]);
    kpart[tid] = s;
  }
  __syncthreads();
  if (tid < 64) ksum[(size_t)bh * 64 + tid] = kpart[tid] + kpart[tid + 64];
}

// ---------------------------------------------------------------------------
// Stage 4: per (b, 128-row tile), loop heads:
//   z = 1/(q . ksum + 1e-6);  y[:,h*64:+64] = (q @ kv_h) * z   (bf16 out)
// q tile staged with async global->LDS (already bf16 in workspace).
// ---------------------------------------------------------------------------
__global__ __launch_bounds__(256) void attn_apply(const u16* __restrict__ q_act,
                                                  const float* __restrict__ kv_mat,
                                                  const float* __restrict__ ksum,
                                                  u16* __restrict__ y_cat) {
  __shared__ __align__(16) u16 qs[128 * 72];   // q[row][d], stride 72 (16B-aligned rows)
  __shared__ u16   kvT[64 * 66];               // kvT[m][d] = kv[d][m]
  __shared__ float ksm[64];
  __shared__ float zpart[256];
  __shared__ float zl[128];

  const int b  = blockIdx.y;
  const int n0 = blockIdx.x * 128;
  const int tid = threadIdx.x, lane = tid & 31, wave = tid >> 5;

  for (int h = 0; h < 6; ++h) {
    const int bh = b * 6 + h;
    {   // q tile: 128x64 bf16, async copy (64B per thread)
      const int r = tid >> 1, db = (tid & 1) << 5;
      const u16* src = q_act + ((size_t)b * 4096 + n0 + r) * 384 + h * 64 + db;
      async_cp_64B(lds_off(qs + r * 72 + db), src);
    }
    {   // kv_h transposed to B-operand layout, f32 -> bf16
      const int m = tid >> 2, db = (tid & 3) << 4;
      const float* src = kv_mat + (size_t)bh * 4096;
#pragma unroll
      for (int i = 0; i < 16; ++i)
        kvT[m * 66 + db + i] = f2bf(src[(db + i) * 64 + m]);
    }
    if (tid < 64) ksm[tid] = ksum[(size_t)bh * 64 + tid];
    wait_async0();
    __syncthreads();

    {   // rowwise z
      const int r = tid >> 1, part = tid & 1;
      float s = 0.0f;
      const u16* qr = qs + r * 72 + part * 32;
#pragma unroll
      for (int i = 0; i < 32; ++i) s += bf2f(qr[i]) * ksm[part * 32 + i];
      zpart[tid] = s;
    }
    __syncthreads();
    if (tid < 128) zl[tid] = 1.0f / (zpart[tid * 2] + zpart[tid * 2 + 1] + 1e-6f);
    __syncthreads();

    // wave w owns rows w*16..+16; 4 col-frags; K=64 -> 2 WMMA steps
    v8f acc[4] = {};
#pragma unroll
    for (int kstep = 0; kstep < 2; ++kstep) {
      v16bf a = lds_frag(qs, wave * 16, 72, kstep * 32, lane);
#pragma unroll
      for (int fn = 0; fn < 4; ++fn) {
        v16bf bb = lds_frag(kvT, fn * 16, 66, kstep * 32, lane);
        acc[fn] = wmma_bf16(a, bb, acc[fn]);
      }
    }
#pragma unroll
    for (int fn = 0; fn < 4; ++fn)
#pragma unroll
      for (int r = 0; r < 8; ++r) {
        const int row = wave * 16 + ((lane >> 4) << 3) + r;
        const int col = fn * 16 + (lane & 15);
        const float v = acc[fn][r] * zl[row];
        y_cat[((size_t)b * 4096 + n0 + row) * 384 + h * 64 + col] = f2bf(v);
      }
    __syncthreads();
  }
}

// ---------------------------------------------------------------------------
extern "C" void kernel_launch(void* const* d_in, const int* in_sizes, int n_in,
                              void* d_out, int out_size, void* d_ws, size_t ws_size,
                              hipStream_t stream) {
  const float* x_q  = (const float*)d_in[0];   // [8,4096,384]
  const float* x_kv = (const float*)d_in[1];   // [8,4096,384]
  const float* Wq   = (const float*)d_in[2];   // [384,384]
  const float* Wkv  = (const float*)d_in[3];   // [384,768]
  const float* Wout = (const float*)d_in[4];   // [384,384]
  const float* bout = (const float*)d_in[5];   // [384]
  float* out = (float*)d_out;                  // [8,4096,384] f32

  const size_t NR = 8 * 4096;                  // 32768 flattened rows
  u16*   q_act  = (u16*)d_ws;                  // [NR,384] bf16  elu(q)+1
  u16*   kvp    = q_act + NR * 384;            // [NR,768] bf16  (k|v), k has elu+1
  u16*   y_cat  = kvp + NR * 768;              // [NR,384] bf16  attn output
  float* kv_mat = (float*)(y_cat + NR * 384);  // [48,64,64] f32
  float* ksum   = kv_mat + 48 * 4096;          // [48,64] f32
  // total ws use: ~101.5 MB

  const dim3 blk(256);
  gemm384<1, false, false, false><<<dim3(256, 6),  blk, 0, stream>>>(x_q,  Wq,  nullptr, q_act, (int)NR, 384);
  gemm384<2, false, false, false><<<dim3(256, 12), blk, 0, stream>>>(x_kv, Wkv, nullptr, kvp,   (int)NR, 768);
  kv_outer<<<dim3(48), dim3(128), 0, stream>>>(kvp, kv_mat, ksum);
  attn_apply<<<dim3(32, 8), blk, 0, stream>>>(q_act, kv_mat, ksum, y_cat);
  gemm384<0, true, true, true><<<dim3(256, 6), blk, 0, stream>>>(y_cat, Wout, bout, out, (int)NR, 384);
}
